// EquivariantGNN_25752623906908
// MI455X (gfx1250) — matrix operand order repaired and checked by reading, not compiled, not run on hardware
//
#include <hip/hip_runtime.h>
#include <math.h>

typedef __attribute__((ext_vector_type(16))) _Float16 v16h;
typedef __attribute__((ext_vector_type(8)))  _Float16 v8h;
typedef __attribute__((ext_vector_type(8)))  float    v8f;

#define S_CH 32
#define V_CH 16
#define NRB 6
#define NUMEL 2304
#define NTILES 144            // 2304 / 16
#define N_SPHERE 256

#define SQRT3F      1.7320508075688772f
#define INV_SQRT3F  0.5773502691896258f
#define A_SCALE     0.14433756729740643f   // 1/sqrt(48)
#define INV_S       0.17677669529663687f   // 1/sqrt(32)
#define INV_V       0.25f                  // 1/sqrt(16)
#define PI_F        3.14159265358979323846f

__device__ __forceinline__ float sigmoidf_(float x) { return 1.0f / (1.0f + __expf(-x)); }
__device__ __forceinline__ float siluf_(float x)    { return x * sigmoidf_(x); }

__device__ __forceinline__ v8f wmma_f16(v16h a, v16h b, v8f c) {
  return __builtin_amdgcn_wmma_f32_16x16x32_f16(false, a, false, b, (short)0, c, false, false);
}

// ---------------------------------------------------------------------------
// Embedding: s[n,w] = (x[n,0:3] @ W_in_s)[w] / sqrt(3);  v[n,w,k] = x[n,3+k]*W_in_v[w]
// ---------------------------------------------------------------------------
__global__ void embed_kernel(const float* __restrict__ x,
                             const float* __restrict__ Wins,
                             const float* __restrict__ Winv,
                             float* __restrict__ s, float* __restrict__ v, int N) {
  int tid = blockIdx.x * blockDim.x + threadIdx.x;
  if (tid >= N * S_CH) return;
  int n = tid / S_CH, w = tid % S_CH;
  const float* xr = x + (long)n * 6;
  float acc = 0.f;
  #pragma unroll
  for (int u = 0; u < 3; ++u) acc += xr[u] * Wins[u * S_CH + w];
  s[(long)n * S_CH + w] = acc * (1.0f / SQRT3F);
  if (w < V_CH) {
    float wv = Winv[w];
    float* vr = v + (long)n * (V_CH * 3) + w * 3;
    #pragma unroll
    for (int k = 0; k < 3; ++k) vr[k] = xr[3 + k] * wv;
  }
}

// ---------------------------------------------------------------------------
// Radial basis per edge (layer independent)
// ---------------------------------------------------------------------------
__global__ void rbf_kernel(const float* __restrict__ ed, float* __restrict__ rb, int E) {
  int e = blockIdx.x * blockDim.x + threadIdx.x;
  if (e >= E) return;
  float d = ed[e];
  float cut = (d < 0.35f) ? 0.5f * (__cosf(d * (PI_F / 0.35f)) + 1.0f) : 0.0f;
  const float width = 0.35f / NRB * 0.5f;
  #pragma unroll
  for (int r = 0; r < NRB; ++r) {
    float c = (float)r * (0.35f / (NRB - 1));
    float z = (d - c) / width;
    rb[(long)e * NRB + r] = __expf(-0.5f * z * z) * cut;
  }
}

// ---------------------------------------------------------------------------
// Pack Wr2 (L,32,2304) f32 -> f16 in per-lane WMMA B-fragment order.
// B fragment (32x16 KxN): lane holds col N=lane&15; h=lane>>4: i<8: K=8h+i, i>=8: K=16+8h+(i-8).
// ---------------------------------------------------------------------------
__global__ void pack_wr2_kernel(const float* __restrict__ Wr2, _Float16* __restrict__ out) {
  int tid = blockIdx.x * blockDim.x + threadIdx.x;
  if (tid >= 2 * NTILES * 32 * 16) return;
  int i    = tid & 15;
  int lane = (tid >> 4) & 31;
  int t    = (tid >> 9) % NTILES;
  int l    = tid / (NTILES * 512);
  int n = t * 16 + (lane & 15);
  int k = (i < 8) ? (8 * (lane >> 4) + i) : (16 + 8 * (lane >> 4) + (i - 8));
  out[tid] = (_Float16)Wr2[((long)l * S_CH + k) * NUMEL + n];
}

// ---------------------------------------------------------------------------
// Radial MLP -> h (silu), stored f16 row-major [E][32]
// ---------------------------------------------------------------------------
__global__ void radial_kernel(const float* __restrict__ rb,
                              const float* __restrict__ Wr1,
                              const float* __restrict__ br1,
                              _Float16* __restrict__ h16, int E) {
  int tid = blockIdx.x * blockDim.x + threadIdx.x;
  if (tid >= E * S_CH) return;
  int e = tid / S_CH, w = tid % S_CH;
  float acc = br1[w];
  const float* rr = rb + (long)e * NRB;
  #pragma unroll
  for (int r = 0; r < NRB; ++r) acc += rr[r] * Wr1[r * S_CH + w];
  h16[tid] = (_Float16)siluf_(acc);
}

// ---------------------------------------------------------------------------
// Self-interaction: s_nxt = s@Wsc_s*inv_s ; v_nxt = v contracted with Wsc_v*inv_v
// ---------------------------------------------------------------------------
__global__ void self_mix_kernel(const float* __restrict__ s, const float* __restrict__ v,
                                const float* __restrict__ Wss, const float* __restrict__ Wsv,
                                float* __restrict__ sn, float* __restrict__ vn, int N) {
  int tid = blockIdx.x * blockDim.x + threadIdx.x;
  if (tid >= N * S_CH) return;
  int n = tid / S_CH, w = tid % S_CH;
  const float* sr = s + (long)n * S_CH;
  float acc = 0.f;
  #pragma unroll
  for (int u = 0; u < S_CH; ++u) acc += sr[u] * Wss[u * S_CH + w];
  sn[(long)n * S_CH + w] = acc * INV_S;
  if (w < V_CH) {
    const float* vr = v + (long)n * (V_CH * 3);
    float a0 = 0.f, a1 = 0.f, a2 = 0.f;
    #pragma unroll
    for (int u = 0; u < V_CH; ++u) {
      float wm = Wsv[u * V_CH + w];
      a0 += vr[u * 3 + 0] * wm; a1 += vr[u * 3 + 1] * wm; a2 += vr[u * 3 + 2] * wm;
    }
    float* vo = vn + (long)n * (V_CH * 3) + w * 3;
    vo[0] = a0 * INV_V; vo[1] = a1 * INV_V; vo[2] = a2 * INV_V;
  }
}

// ---------------------------------------------------------------------------
// Fused edge message kernel. One wave per 16-edge tile.
// tpw tile (16x16) = WMMA(h(16x32 f16), Wr2 tile(32x16 f16)) + br2, consumed
// in-register against per-edge scalars staged in LDS. 144 WMMAs per wave,
// branchless: tiles 0..95 (W1|W2 scalar path), tiles 96..143 (W3|W4 vector path).
// ---------------------------------------------------------------------------
__global__ __launch_bounds__(32) void edge_msg_kernel(
    const float* __restrict__ s, const float* __restrict__ v,
    const int* __restrict__ edge_index, const float* __restrict__ edge_sh,
    const _Float16* __restrict__ h16, const _Float16* __restrict__ wr2p,
    const float* __restrict__ br2,
    float* __restrict__ msgS, float* __restrict__ msgV, int E) {
  __shared__ float scalL[16][48];      // [m][u]: u<32 -> s_j*y0 ; u>=32 -> (v_j.y1)/sqrt3
  __shared__ float vscalL[16][48][3];  // [m][u][k]: u<32 -> s_j[u]*y1[k] ; u>=32 -> v_j[u-32][k]*y0
  __shared__ float br2L[NUMEL];        // bias slice for this layer

  const int lane = threadIdx.x;
  const int half = lane >> 4;
  const int ln   = lane & 15;
  const long e0  = (long)blockIdx.x * 16;

  if (lane < 16) {
    long e = e0 + lane; if (e >= E) e = E - 1;
    int src = edge_index[e];
    float y0  = edge_sh[e * 4 + 0];
    float y1a = edge_sh[e * 4 + 1], y1b = edge_sh[e * 4 + 2], y1c = edge_sh[e * 4 + 3];
    const float* sr = s + (long)src * S_CH;
    #pragma unroll
    for (int u = 0; u < S_CH; ++u) {
      float sv = sr[u];
      scalL[lane][u] = sv * y0;
      vscalL[lane][u][0] = sv * y1a; vscalL[lane][u][1] = sv * y1b; vscalL[lane][u][2] = sv * y1c;
    }
    const float* vr = v + (long)src * (V_CH * 3);
    #pragma unroll
    for (int u = 0; u < V_CH; ++u) {
      float a0 = vr[u * 3 + 0], a1 = vr[u * 3 + 1], a2 = vr[u * 3 + 2];
      scalL[lane][32 + u] = INV_SQRT3F * (a0 * y1a + a1 * y1b + a2 * y1c);
      vscalL[lane][32 + u][0] = a0 * y0; vscalL[lane][32 + u][1] = a1 * y0; vscalL[lane][32 + u][2] = a2 * y0;
    }
  } else {
    for (int idx = lane - 16; idx < NUMEL; idx += 16) br2L[idx] = br2[idx];
  }
  __syncthreads();

  // A fragment: row M = lane&15; K-map = two contiguous runs of 8 -> two b128 loads.
  v16h afrag;
  {
    long eA = e0 + ln; if (eA >= E) eA = E - 1;
    const _Float16* hr = h16 + eA * S_CH + 8 * half;
    v8h lo = *(const v8h*)hr;          // K = 8*half .. 8*half+7
    v8h hi = *(const v8h*)(hr + 16);   // K = 8*half+16 .. 8*half+23
    afrag = __builtin_shufflevector(lo, hi, 0,1,2,3,4,5,6,7,8,9,10,11,12,13,14,15);
  }

  float msS0[8], msS1[8], mv0[8], mv1[8], mv2[8];
  #pragma unroll
  for (int j = 0; j < 8; ++j) { msS0[j]=0.f; msS1[j]=0.f; mv0[j]=0.f; mv1[j]=0.f; mv2[j]=0.f; }

  const _Float16* bptr = wr2p + (long)lane * 16;  // tile stride = 512 f16

  // ---- W1 (tiles 0..63) + W2 (tiles 64..95): scalar-message path ----
  // tile pair (2tp, 2tp+1) shares scalar index tp; even tile -> w, odd -> w+16
  for (int tp = 0; tp < 48; ++tp) {
    const int t0 = tp * 2;
    v16h b0 = *(const v16h*)(bptr + (long)t0 * 512);
    v16h b1 = *(const v16h*)(bptr + (long)t0 * 512 + 512);
    float bias0 = br2L[t0 * 16 + ln];
    float bias1 = br2L[t0 * 16 + 16 + ln];
    v8f c0, c1;
    #pragma unroll
    for (int j = 0; j < 8; ++j) { c0[j] = bias0; c1[j] = bias1; }
    c0 = wmma_f16(afrag, b0, c0);
    c1 = wmma_f16(afrag, b1, c1);
    #pragma unroll
    for (int j = 0; j < 8; ++j) {
      const int M = j + 8 * half;
      float sc = scalL[M][tp];
      msS0[j] += sc * c0[j];
      msS1[j] += sc * c1[j];
    }
  }

  // ---- W3 (tiles 96..127) + W4 (tiles 128..143): vector-message path ----
  for (int t = 96; t < NTILES; ++t) {
    v16h b = *(const v16h*)(bptr + (long)t * 512);
    float bias = br2L[t * 16 + ln];
    v8f c;
    #pragma unroll
    for (int j = 0; j < 8; ++j) c[j] = bias;
    c = wmma_f16(afrag, b, c);
    const int u = t - 96;
    #pragma unroll
    for (int j = 0; j < 8; ++j) {
      const int M = j + 8 * half;
      mv0[j] += vscalL[M][u][0] * c[j];
      mv1[j] += vscalL[M][u][1] * c[j];
      mv2[j] += vscalL[M][u][2] * c[j];
    }
  }

  #pragma unroll
  for (int j = 0; j < 8; ++j) {
    const int M = j + 8 * half;
    long e = e0 + M;
    if (e < E) {
      msgS[e * S_CH + ln]      = A_SCALE * msS0[j];
      msgS[e * S_CH + 16 + ln] = A_SCALE * msS1[j];
      float* mv = msgV + e * (V_CH * 3) + ln * 3;
      mv[0] = A_SCALE * mv0[j]; mv[1] = A_SCALE * mv1[j]; mv[2] = A_SCALE * mv2[j];
    }
  }
}

// ---------------------------------------------------------------------------
// Per-edge gating + linear-out + scatter-add into node accumulators
// ---------------------------------------------------------------------------
__global__ void gate_scatter_kernel(const float* __restrict__ msgS, const float* __restrict__ msgV,
                                    const int* __restrict__ edge_index,
                                    const float* __restrict__ Wpgs, const float* __restrict__ Wpgv,
                                    const float* __restrict__ Wlos, const float* __restrict__ Wlov,
                                    float* __restrict__ sn, float* __restrict__ vn, int E) {
  int e = blockIdx.x * blockDim.x + threadIdx.x;
  if (e >= E) return;
  float ms[S_CH];
  const float* mr = msgS + (long)e * S_CH;
  #pragma unroll
  for (int u = 0; u < S_CH; ++u) ms[u] = mr[u];

  float g_s[S_CH], gate[V_CH];
  #pragma unroll
  for (int j = 0; j < S_CH + V_CH; ++j) {
    float acc = 0.f;
    #pragma unroll
    for (int u = 0; u < S_CH; ++u) acc += ms[u] * Wpgs[u * (S_CH + V_CH) + j];
    acc *= INV_S;
    if (j < S_CH) g_s[j] = siluf_(acc); else gate[j - S_CH] = sigmoidf_(acc);
  }

  float mv[V_CH * 3];
  const float* mvr = msgV + (long)e * (V_CH * 3);
  #pragma unroll
  for (int u = 0; u < V_CH * 3; ++u) mv[u] = mvr[u];

  float gv0[V_CH], gv1[V_CH], gv2[V_CH];
  #pragma unroll
  for (int w = 0; w < V_CH; ++w) {
    float a0 = 0.f, a1 = 0.f, a2 = 0.f;
    #pragma unroll
    for (int u = 0; u < V_CH; ++u) {
      float wm = Wpgv[u * V_CH + w];
      a0 += mv[u * 3 + 0] * wm; a1 += mv[u * 3 + 1] * wm; a2 += mv[u * 3 + 2] * wm;
    }
    float g = gate[w] * INV_V;
    gv0[w] = g * a0; gv1[w] = g * a1; gv2[w] = g * a2;
  }

  int dst = edge_index[(long)E + e];
  float* srow = sn + (long)dst * S_CH;
  float* vrow = vn + (long)dst * (V_CH * 3);
  #pragma unroll
  for (int w = 0; w < S_CH; ++w) {
    float acc = 0.f;
    #pragma unroll
    for (int u = 0; u < S_CH; ++u) acc += g_s[u] * Wlos[u * S_CH + w];
    atomicAdd(&srow[w], acc * INV_S);
  }
  #pragma unroll
  for (int w = 0; w < V_CH; ++w) {
    float a0 = 0.f, a1 = 0.f, a2 = 0.f;
    #pragma unroll
    for (int u = 0; u < V_CH; ++u) {
      float wm = Wlov[u * V_CH + w];
      a0 += gv0[u] * wm; a1 += gv1[u] * wm; a2 += gv2[u] * wm;
    }
    atomicAdd(&vrow[w * 3 + 0], a0 * INV_V);
    atomicAdd(&vrow[w * 3 + 1], a1 * INV_V);
    atomicAdd(&vrow[w * 3 + 2], a2 * INV_V);
  }
}

// ---------------------------------------------------------------------------
// M[r][k] = (1/256) sum_p Y[p,r]*dirs[p,k], Y = [1, sqrt3*dirs]
// ---------------------------------------------------------------------------
__global__ void sphere_mat_kernel(const float* __restrict__ dirs, float* __restrict__ M43) {
  int tid = threadIdx.x;
  if (tid >= 12) return;
  int r = tid / 3, k = tid % 3;
  float acc = 0.f;
  for (int p = 0; p < N_SPHERE; ++p) {
    float y = (r == 0) ? 1.0f : SQRT3F * dirs[p * 3 + (r - 1)];
    acc += y * dirs[p * 3 + k];
  }
  M43[r * 3 + k] = acc * (1.0f / N_SPHERE);
}

__global__ void out_kernel(const float* __restrict__ s, const float* __restrict__ v,
                           const float* __restrict__ Wouts, const float* __restrict__ Woutv,
                           const float* __restrict__ M43, float* __restrict__ out, int N) {
  int n = blockIdx.x * blockDim.x + threadIdx.x;
  if (n >= N) return;
  const float* sr = s + (long)n * S_CH;
  float c0 = 0.f;
  #pragma unroll
  for (int u = 0; u < S_CH; ++u) c0 += sr[u] * Wouts[u];
  c0 *= INV_S;
  const float* vr = v + (long)n * (V_CH * 3);
  float cv[3] = {0.f, 0.f, 0.f};
  #pragma unroll
  for (int u = 0; u < V_CH; ++u) {
    float wv = Woutv[u];
    cv[0] += vr[u * 3 + 0] * wv; cv[1] += vr[u * 3 + 1] * wv; cv[2] += vr[u * 3 + 2] * wv;
  }
  #pragma unroll
  for (int k = 0; k < 3; ++k) {
    float o = c0 * M43[0 * 3 + k];
    o += cv[0] * INV_V * M43[1 * 3 + k];
    o += cv[1] * INV_V * M43[2 * 3 + k];
    o += cv[2] * INV_V * M43[3 * 3 + k];
    out[(long)n * 3 + k] = o;
  }
}

// ---------------------------------------------------------------------------
extern "C" void kernel_launch(void* const* d_in, const int* in_sizes, int n_in,
                              void* d_out, int out_size, void* d_ws, size_t ws_size,
                              hipStream_t stream) {
  const float* x      = (const float*)d_in[0];
  const int*   eidx   = (const int*)  d_in[1];
  const float* edist  = (const float*)d_in[2];
  const float* esh    = (const float*)d_in[3];
  const float* dirs   = (const float*)d_in[4];
  const float* Wins   = (const float*)d_in[5];
  const float* Winv   = (const float*)d_in[6];
  const float* Wscs   = (const float*)d_in[7];
  const float* Wscv   = (const float*)d_in[8];
  const float* Wr1    = (const float*)d_in[9];
  const float* br1    = (const float*)d_in[10];
  const float* Wr2    = (const float*)d_in[11];
  const float* br2    = (const float*)d_in[12];
  const float* Wpgs   = (const float*)d_in[13];
  const float* Wpgv   = (const float*)d_in[14];
  const float* Wlos   = (const float*)d_in[15];
  const float* Wlov   = (const float*)d_in[16];
  const float* Wouts  = (const float*)d_in[17];
  const float* Woutv  = (const float*)d_in[18];
  float* out = (float*)d_out;

  const int N = in_sizes[0] / 6;
  const int E = in_sizes[2];

  // workspace layout
  float* sA   = (float*)d_ws;
  float* sB   = sA + (long)N * S_CH;
  float* vA   = sB + (long)N * S_CH;
  float* vB   = vA + (long)N * V_CH * 3;
  float* rb   = vB + (long)N * V_CH * 3;
  float* msgS = rb + (long)E * NRB;
  float* msgV = msgS + (long)E * S_CH;
  float* M43  = msgV + (long)E * V_CH * 3;
  _Float16* h16  = (_Float16*)(M43 + 16);
  _Float16* wr2p = h16 + (long)E * S_CH;
  (void)ws_size; (void)n_in; (void)out_size;

  const int TB = 256;
  embed_kernel<<<(N * S_CH + TB - 1) / TB, TB, 0, stream>>>(x, Wins, Winv, sA, vA, N);
  rbf_kernel<<<(E + TB - 1) / TB, TB, 0, stream>>>(edist, rb, E);
  pack_wr2_kernel<<<(2 * NTILES * 32 * 16 + TB - 1) / TB, TB, 0, stream>>>(Wr2, wr2p);

  float* scur = sA; float* vcur = vA;
  float* snxt = sB; float* vnxt = vB;
  for (int l = 0; l < 2; ++l) {
    radial_kernel<<<(E * S_CH + TB - 1) / TB, TB, 0, stream>>>(
        rb, Wr1 + (long)l * NRB * S_CH, br1 + (long)l * S_CH, h16, E);
    self_mix_kernel<<<(N * S_CH + TB - 1) / TB, TB, 0, stream>>>(
        scur, vcur, Wscs + (long)l * S_CH * S_CH, Wscv + (long)l * V_CH * V_CH, snxt, vnxt, N);
    edge_msg_kernel<<<(E + 15) / 16, 32, 0, stream>>>(
        scur, vcur, eidx, esh, h16, wr2p + (long)l * NTILES * 512,
        br2 + (long)l * NUMEL, msgS, msgV, E);
    gate_scatter_kernel<<<(E + TB - 1) / TB, TB, 0, stream>>>(
        msgS, msgV, eidx,
        Wpgs + (long)l * S_CH * (S_CH + V_CH), Wpgv + (long)l * V_CH * V_CH,
        Wlos + (long)l * S_CH * S_CH, Wlov + (long)l * V_CH * V_CH,
        snxt, vnxt, E);
    float* ts = scur; scur = snxt; snxt = ts;
    float* tv = vcur; vcur = vnxt; vnxt = tv;
  }

  sphere_mat_kernel<<<1, 32, 0, stream>>>(dirs, M43);
  out_kernel<<<(N + TB - 1) / TB, TB, 0, stream>>>(scur, vcur, Wouts, Woutv, M43, out, N);
}
